// TempAjcnMemN2N_21680994910435
// MI455X (gfx1250) — compile-verified, matrix operand order checked by reading, not run on hardware
//
#include <hip/hip_runtime.h>
#include <math.h>

// Problem constants (from reference)
#define MEMN  50
#define HID   256
#define EMB   256
#define BATCH 128
#define SEQ   50
#define ROWS  (MEMN * BATCH)   // 6400 rows of x_sum

typedef __attribute__((ext_vector_type(16))) __bf16 v16bf;
typedef __attribute__((ext_vector_type(8)))  float  v8f;
typedef __attribute__((ext_vector_type(4)))  float  v4f;

static __device__ __forceinline__ unsigned short f32_to_bf16(float f) {
  unsigned int u = __float_as_uint(f);
  u += 0x7FFFu + ((u >> 16) & 1u);   // round-to-nearest-even
  return (unsigned short)(u >> 16);
}

// ---------------------------------------------------------------------------
// Kernel 1: sum over LEN. 327.7 MB read-once stream -> x_sum/q_sum in bf16.
// Each 256-thread block covers 4 rows; 64 lanes * v4f = one 256-ch row.
// Per wave-instruction: 512B contiguous non-temporal global_load_b128.
// ---------------------------------------------------------------------------
__global__ void reduce_len_kernel(const float* __restrict__ inputs,
                                  const float* __restrict__ questions,
                                  unsigned short* __restrict__ xsum_bf,
                                  unsigned short* __restrict__ qsum_bf) {
  const int NBIG = ROWS / 4;            // 1600 blocks for inputs
  int sub = threadIdx.x >> 6;           // row within block (0..3)
  int ch4 = (threadIdx.x & 63) * 4;     // channel group
  v4f s = {0.f, 0.f, 0.f, 0.f};
  unsigned short* dst;
  if (blockIdx.x < NBIG) {
    int row = blockIdx.x * 4 + sub;     // row = m*BATCH + b
    const float* src = inputs + ((size_t)row * SEQ) * EMB + ch4;
    dst = xsum_bf + (size_t)row * EMB + ch4;
#pragma unroll
    for (int l = 0; l < SEQ; ++l)
      s += __builtin_nontemporal_load((const v4f*)(src + (size_t)l * EMB));
  } else {
    int b = (blockIdx.x - NBIG) * 4 + sub;
    const float* src = questions + ((size_t)b * SEQ) * EMB + ch4;
    dst = qsum_bf + (size_t)b * EMB + ch4;
#pragma unroll
    for (int l = 0; l < SEQ; ++l)
      s += *(const v4f*)(src + (size_t)l * EMB);
  }
  union { unsigned short h[4]; uint2 u2; } pk;
  pk.h[0] = f32_to_bf16(s.x); pk.h[1] = f32_to_bf16(s.y);
  pk.h[2] = f32_to_bf16(s.z); pk.h[3] = f32_to_bf16(s.w);
  *(uint2*)dst = pk.u2;
}

// ---------------------------------------------------------------------------
// Kernel 2: convert 5 weight matrices [EMB(k) x HID(h)] f32 -> bf16,
// transposed to Bt[h][k] so WMMA B-operand K-pairs are contiguous.
// g order: 0=A_1, 1=weights[0], 2=weights[1], 3=C_k, 4=B
// ---------------------------------------------------------------------------
__global__ void prep_weights_kernel(const float* __restrict__ A_1,
                                    const float* __restrict__ C_k,
                                    const float* __restrict__ B,
                                    const float* __restrict__ weights,
                                    unsigned short* __restrict__ bt) {
  int g = blockIdx.x >> 8;    // 0..4
  int c = blockIdx.x & 255;   // output channel h
  int k = threadIdx.x;        // 0..255
  const float* W;
  switch (g) {
    case 0:  W = A_1; break;
    case 1:  W = weights; break;
    case 2:  W = weights + (size_t)EMB * HID; break;
    case 3:  W = C_k; break;
    default: W = B; break;
  }
  bt[((size_t)g * HID + c) * EMB + k] = f32_to_bf16(W[(size_t)k * HID + c]);
}

// ---------------------------------------------------------------------------
// Kernel 3: WMMA GEMMs. proj[g] = x_sum @ W_g (g=0..3), u0 = q_sum @ B.
// One wave per 16x16 tile; K loop = 8 steps of 32 (v_wmma_f32_16x16x32_bf16).
// A-operand lane layout (ISA 7.12.2, 16-bit A 16x32): per lane, K-pairs
// {0,2,4,6}+8*half and {16,18,20,22}+8*half -> two contiguous 16B chunks,
// which the compiler merges into two global_load_b128 per operand.
// ---------------------------------------------------------------------------
__global__ void wmma_gemm_kernel(const unsigned short* __restrict__ xsum_bf,
                                 const unsigned short* __restrict__ qsum_bf,
                                 const unsigned short* __restrict__ bt,
                                 float* __restrict__ proj,
                                 float* __restrict__ u0) {
  const int TILES_BIG = (ROWS / 16) * (HID / 16);  // 6400 tiles per projection
  const int TOTAL_BIG = 4 * TILES_BIG;             // 25600
  int wave = blockIdx.x * (blockDim.x >> 5) + (threadIdx.x >> 5);
  int lane = threadIdx.x & 31;

  const unsigned short* A;
  const unsigned short* Bm;
  float* out;
  int mtile, ntile;
  if (wave < TOTAL_BIG) {
    int g = wave / TILES_BIG;
    int r = wave - g * TILES_BIG;
    mtile = r >> 4;            // 400 M-tiles
    ntile = r & 15;            // 16 N-tiles
    A   = xsum_bf;
    Bm  = bt + (size_t)g * EMB * HID;
    out = proj + (size_t)g * ROWS * HID;
  } else {
    int r = wave - TOTAL_BIG;  // 0..127 : u0 = q_sum @ B
    mtile = r >> 4;
    ntile = r & 15;
    A   = qsum_bf;
    Bm  = bt + (size_t)4 * EMB * HID;
    out = u0;
  }

  int half = lane >> 4;        // selects K sub-range
  int lm   = lane & 15;        // M (for A) / N (for B) within tile
  const unsigned short* arow = A  + (size_t)(mtile * 16 + lm) * EMB;
  const unsigned short* brow = Bm + (size_t)(ntile * 16 + lm) * EMB;

  union { unsigned int u[8]; v16bf v; } a, b;
  v8f acc = {};
#pragma unroll 1
  for (int k0 = 0; k0 < EMB; k0 += 32) {
#pragma unroll
    for (int p = 0; p < 8; ++p) {
      // K-pair base: p<4 -> 2p ; p>=4 -> 16+2(p-4) ; +8 for upper half-wave
      int kb = k0 + ((p < 4) ? (p * 2) : (8 + p * 2)) + half * 8;
      a.u[p] = *(const unsigned int*)(arow + kb);   // packs K, K+1 (lo, hi)
      b.u[p] = *(const unsigned int*)(brow + kb);
    }
    acc = __builtin_amdgcn_wmma_f32_16x16x32_bf16(
        /*neg_a=*/false, a.v, /*neg_b=*/false, b.v,
        /*c_mod=*/(short)0, acc, /*reuse_a=*/false, /*reuse_b=*/false);
  }

  // C/D layout: VGPR v -> row = v + 8*half ; col = lane&15
  int col = ntile * 16 + lm;
#pragma unroll
  for (int v = 0; v < 8; ++v) {
    int row = mtile * 16 + v + half * 8;
    out[(size_t)row * HID + col] = acc[v];
  }
}

// ---------------------------------------------------------------------------
// Kernel 4: fused 3-hop attention. One block per batch element, u in LDS.
// hop i uses mem_m = proj[i] + TA_i, mem_c = proj[i+1] + TC_i.
// ---------------------------------------------------------------------------
__global__ void hops_kernel(const float* __restrict__ proj,
                            const float* __restrict__ u0,
                            const float* __restrict__ TA_1,
                            const float* __restrict__ TC_k,
                            const float* __restrict__ T_weights,
                            float* __restrict__ out) {
  __shared__ float u_s[HID];
  __shared__ float ls[MEMN];
  __shared__ float ps[MEMN];
  int b = blockIdx.x, tid = threadIdx.x;
  u_s[tid] = u0[(size_t)b * HID + tid];
  __syncthreads();

  for (int hop = 0; hop < 3; ++hop) {
    const float* pm = proj + (size_t)hop * ROWS * HID;
    const float* pc = proj + (size_t)(hop + 1) * ROWS * HID;
    const float* TA = (hop == 0) ? TA_1 : (T_weights + (size_t)(hop - 1) * MEMN * HID);
    const float* TC = (hop == 2) ? TC_k : (T_weights + (size_t)hop * MEMN * HID);

    if (tid < MEMN) {  // logits[m] = sum_h u[h] * (proj_m[m,b,h] + TA[m,h])
      const v4f* row  = (const v4f*)(pm + ((size_t)tid * BATCH + b) * HID);
      const v4f* trow = (const v4f*)(TA + (size_t)tid * HID);
      const v4f* us4  = (const v4f*)u_s;
      float s = 0.f;
#pragma unroll 4
      for (int h = 0; h < HID / 4; ++h) {
        v4f r = row[h], t = trow[h], u = us4[h];
        s += u.x * (r.x + t.x) + u.y * (r.y + t.y) +
             u.z * (r.z + t.z) + u.w * (r.w + t.w);
      }
      ls[tid] = s;
    }
    __syncthreads();

    float mx = -1e30f;
    for (int m = 0; m < MEMN; ++m) mx = fmaxf(mx, ls[m]);
    if (tid < MEMN) ps[tid] = expf(ls[tid] - mx);
    __syncthreads();

    float sum = 0.f;
    for (int m = 0; m < MEMN; ++m) sum += ps[m];

    float o = 0.f;   // o[h] = sum_m p[m] * (proj_c[m,b,h] + TC[m,h])
    for (int m = 0; m < MEMN; ++m)
      o += ps[m] * (pc[((size_t)m * BATCH + b) * HID + tid] + TC[(size_t)m * HID + tid]);

    u_s[tid] += o / sum;
    __syncthreads();
  }
  out[(size_t)b * HID + tid] = u_s[tid];
}

// ---------------------------------------------------------------------------
extern "C" void kernel_launch(void* const* d_in, const int* in_sizes, int n_in,
                              void* d_out, int out_size, void* d_ws, size_t ws_size,
                              hipStream_t stream) {
  const float* inputs    = (const float*)d_in[0];
  const float* questions = (const float*)d_in[1];
  const float* A_1       = (const float*)d_in[2];
  const float* C_k       = (const float*)d_in[3];
  const float* B         = (const float*)d_in[4];
  const float* TA_1      = (const float*)d_in[5];
  const float* TC_k      = (const float*)d_in[6];
  const float* weights   = (const float*)d_in[7];
  const float* T_weights = (const float*)d_in[8];
  float* out = (float*)d_out;

  // Workspace layout (~30.3 MB)
  char* ws = (char*)d_ws;
  unsigned short* xsum_bf = (unsigned short*)ws;                 // 6400*256 bf16
  ws += (size_t)ROWS * EMB * sizeof(unsigned short);
  unsigned short* qsum_bf = (unsigned short*)ws;                 // 128*256 bf16
  ws += (size_t)BATCH * EMB * sizeof(unsigned short);
  unsigned short* bt = (unsigned short*)ws;                      // 5*256*256 bf16 (transposed)
  ws += (size_t)5 * EMB * HID * sizeof(unsigned short);
  float* proj = (float*)ws;                                      // 4*6400*256 f32
  ws += (size_t)4 * ROWS * HID * sizeof(float);
  float* u0 = (float*)ws;                                        // 128*256 f32
  (void)ws_size; (void)in_sizes; (void)n_in; (void)out_size;

  // 1) LEN reduction (memory-bound, ~14us HBM roofline), b128 streaming loads
  reduce_len_kernel<<<ROWS / 4 + BATCH / 4, 256, 0, stream>>>(inputs, questions,
                                                              xsum_bf, qsum_bf);
  // 2) weight convert + transpose (tiny)
  prep_weights_kernel<<<5 * 256, 256, 0, stream>>>(A_1, C_k, B, weights, bt);
  // 3) all GEMMs via bf16 WMMA: 25728 tiles / 8 waves-per-block = 3216 blocks exact
  wmma_gemm_kernel<<<3216, 256, 0, stream>>>(xsum_bf, qsum_bf, bt, proj, u0);
  // 4) fused 3-hop attention
  hops_kernel<<<BATCH, 256, 0, stream>>>(proj, u0, TA_1, TC_k, T_weights, out);
}
